// GINEncoder_75771813036514
// MI455X (gfx1250) — matrix operand order repaired and checked by reading, not compiled, not run on hardware
//
#include <hip/hip_runtime.h>

typedef __attribute__((ext_vector_type(16))) _Float16 v16h;
typedef __attribute__((ext_vector_type(8)))  float    v8f;

#define N_NODES 50000
#define N_EDGES 800000
#define DIM     64
#define NLAYERS 5
#define NGRAPH  512
#define NOUT    768

// ---------------------------------------------------------------------------
// Pack f32 weights [K=64][ncols] into f16 WMMA B-fragment layout:
//   P[((nt*2+kt)*32 + lane)*16 + e] = W[(kt*32 + (lane<16?0:8) + (e&7) + (e&8?16:0))*ncols + nt*16 + (lane&15)]
// so each lane's fragment is 16 contiguous halves (32B) -> 2x b128 loads.
// ---------------------------------------------------------------------------
__global__ void pack_b_kernel(const float* __restrict__ W, _Float16* __restrict__ P,
                              int ncols, int nmat) {
    int idx = blockIdx.x * blockDim.x + threadIdx.x;
    int per = DIM * ncols;
    if (idx >= per * nmat) return;
    int mat = idx / per;
    int r   = idx - mat * per;
    int e    = r & 15;
    int lane = (r >> 4) & 31;
    int kt   = (r >> 9) & 1;
    int nt   = r >> 10;
    int K    = kt * 32 + ((lane < 16) ? 0 : 8) + (e & 7) + ((e & 8) ? 16 : 0);
    int col  = nt * 16 + (lane & 15);
    P[idx] = (_Float16)W[mat * per + K * ncols + col];
}

// ---------------------------------------------------------------------------
// Edge scatter-add: agg[dst] += h[src].  1 thread per (edge, float4 chunk).
// ---------------------------------------------------------------------------
__global__ void scatter_kernel(const float* __restrict__ h,
                               const int* __restrict__ esrc,
                               const int* __restrict__ edst,
                               float* __restrict__ agg) {
    long long t = (long long)blockIdx.x * blockDim.x + threadIdx.x;
    if (t >= (long long)N_EDGES * 16) return;
    int e = (int)(t >> 4);
    int q = (int)(t & 15);
    int s = esrc[e], d = edst[e];
    const float4 v = ((const float4*)h)[(long long)s * 16 + q];
    float* p = agg + (long long)d * DIM + q * 4;
    atomicAdd(p + 0, v.x);
    atomicAdd(p + 1, v.y);
    atomicAdd(p + 2, v.z);
    atomicAdd(p + 3, v.w);
}

// ---------------------------------------------------------------------------
// Fused GIN layer: out = mlp((1+eps)*h + agg).  64 nodes/block, 128 threads.
// Wave w owns rows [16w,16w+16): barrier-free.  All 8 B-fragments of each
// GEMM are hoisted into registers, then 8 WMMAs issue back-to-back.
// ---------------------------------------------------------------------------
__global__ void __launch_bounds__(128)
gin_mlp_kernel(const float* __restrict__ h, const float* __restrict__ agg,
               const _Float16* __restrict__ w1p, const _Float16* __restrict__ w2p,
               const float* __restrict__ b1, const float* __restrict__ b2,
               const float* __restrict__ eps, int apply_relu,
               float* __restrict__ out) {
    __shared__ __align__(32) _Float16 sZ[64 * DIM];
    __shared__ __align__(32) _Float16 sT[64 * DIM];

    const int tid  = threadIdx.x;
    const int wave = tid >> 5;
    const int lane = tid & 31;
    const int rowBase = blockIdx.x * 64;

    // stage z = (1+eps)*h + agg as f16; each thread: 32 contiguous elems (half row)
    {
        const float ep = 1.0f + eps[0];
        const int base = tid * 32;
        const int row  = rowBase + (base >> 6);
        const int colb = base & 63;
        if (row < N_NODES) {
            const float4* hp = (const float4*)(h   + (long long)row * DIM + colb);
            const float4* ap = (const float4*)(agg + (long long)row * DIM + colb);
#pragma unroll
            for (int i = 0; i < 8; ++i) {
                float4 hv = hp[i], av = ap[i];
                sZ[base + i * 4 + 0] = (_Float16)(ep * hv.x + av.x);
                sZ[base + i * 4 + 1] = (_Float16)(ep * hv.y + av.y);
                sZ[base + i * 4 + 2] = (_Float16)(ep * hv.z + av.z);
                sZ[base + i * 4 + 3] = (_Float16)(ep * hv.w + av.w);
            }
        } else {
#pragma unroll
            for (int i = 0; i < 32; ++i) sZ[base + i] = (_Float16)0.f;
        }
    }

    const int m  = lane & 15;
    const int ko = (lane < 16) ? 0 : 8;
    const int n  = lane & 15;
    const int mo = (lane < 16) ? 0 : 8;

    auto loadA = [&](const _Float16* S, int kt) {
        v16h a;
#pragma unroll
        for (int e = 0; e < 8; ++e) {
            a[e]     = S[m * DIM + kt * 32 + ko + e];
            a[e + 8] = S[m * DIM + kt * 32 + 16 + ko + e];
        }
        return a;
    };

    const v16h* B1 = (const v16h*)w1p;
    const v16h* B2 = (const v16h*)w2p;

    // ---- GEMM1: hoist all 8 B fragments, then 8 WMMAs back-to-back ----
    const _Float16* Az = sZ + wave * 16 * DIM;
    v16h a0 = loadA(Az, 0);
    v16h a1 = loadA(Az, 1);
    v16h b1f[8];
#pragma unroll
    for (int f = 0; f < 8; ++f) b1f[f] = B1[f * 32 + lane];   // f = nt*2 + kt

    v8f c0 = {}, c1 = {}, c2 = {}, c3 = {};
    c0 = __builtin_amdgcn_wmma_f32_16x16x32_f16(false, a0, false, b1f[0], (short)0, c0, false, false);
    c1 = __builtin_amdgcn_wmma_f32_16x16x32_f16(false, a0, false, b1f[2], (short)0, c1, false, false);
    c2 = __builtin_amdgcn_wmma_f32_16x16x32_f16(false, a0, false, b1f[4], (short)0, c2, false, false);
    c3 = __builtin_amdgcn_wmma_f32_16x16x32_f16(false, a0, false, b1f[6], (short)0, c3, false, false);
    c0 = __builtin_amdgcn_wmma_f32_16x16x32_f16(false, a1, false, b1f[1], (short)0, c0, false, false);
    c1 = __builtin_amdgcn_wmma_f32_16x16x32_f16(false, a1, false, b1f[3], (short)0, c1, false, false);
    c2 = __builtin_amdgcn_wmma_f32_16x16x32_f16(false, a1, false, b1f[5], (short)0, c2, false, false);
    c3 = __builtin_amdgcn_wmma_f32_16x16x32_f16(false, a1, false, b1f[7], (short)0, c3, false, false);

    {
        v8f cc[4] = {c0, c1, c2, c3};
#pragma unroll
        for (int nt = 0; nt < 4; ++nt) {
            float bias = b1[nt * 16 + n];
#pragma unroll
            for (int r = 0; r < 8; ++r) {
                float v = cc[nt][r] + bias;
                v = v > 0.f ? v : 0.f;
                sT[(wave * 16 + mo + r) * DIM + nt * 16 + n] = (_Float16)v;
            }
        }
    }

    // ---- GEMM2: same pattern on this wave's sT rows ----
    const _Float16* At = sT + wave * 16 * DIM;
    v16h t0 = loadA(At, 0);
    v16h t1 = loadA(At, 1);
    v16h b2f[8];
#pragma unroll
    for (int f = 0; f < 8; ++f) b2f[f] = B2[f * 32 + lane];

    v8f d0 = {}, d1 = {}, d2 = {}, d3 = {};
    d0 = __builtin_amdgcn_wmma_f32_16x16x32_f16(false, t0, false, b2f[0], (short)0, d0, false, false);
    d1 = __builtin_amdgcn_wmma_f32_16x16x32_f16(false, t0, false, b2f[2], (short)0, d1, false, false);
    d2 = __builtin_amdgcn_wmma_f32_16x16x32_f16(false, t0, false, b2f[4], (short)0, d2, false, false);
    d3 = __builtin_amdgcn_wmma_f32_16x16x32_f16(false, t0, false, b2f[6], (short)0, d3, false, false);
    d0 = __builtin_amdgcn_wmma_f32_16x16x32_f16(false, t1, false, b2f[1], (short)0, d0, false, false);
    d1 = __builtin_amdgcn_wmma_f32_16x16x32_f16(false, t1, false, b2f[3], (short)0, d1, false, false);
    d2 = __builtin_amdgcn_wmma_f32_16x16x32_f16(false, t1, false, b2f[5], (short)0, d2, false, false);
    d3 = __builtin_amdgcn_wmma_f32_16x16x32_f16(false, t1, false, b2f[7], (short)0, d3, false, false);

    {
        v8f dd[4] = {d0, d1, d2, d3};
#pragma unroll
        for (int nt = 0; nt < 4; ++nt) {
            float bias = b2[nt * 16 + n];
#pragma unroll
            for (int r = 0; r < 8; ++r) {
                float v = dd[nt][r] + bias;
                if (apply_relu) v = v > 0.f ? v : 0.f;
                int row = rowBase + wave * 16 + mo + r;
                if (row < N_NODES)
                    out[(long long)row * DIM + nt * 16 + n] = v;
            }
        }
    }
}

// ---------------------------------------------------------------------------
// Mean pool: atomic accumulation into per-graph sums/counts (float4/thread)
// ---------------------------------------------------------------------------
__global__ void pool_scatter_kernel(const float* __restrict__ h,
                                    const int* __restrict__ batch,
                                    float* __restrict__ sums,
                                    float* __restrict__ cnts) {
    long long t = (long long)blockIdx.x * blockDim.x + threadIdx.x;
    if (t >= (long long)N_NODES * 16) return;
    int node = (int)(t >> 4);
    int q    = (int)(t & 15);
    int g = batch[node];
    const float4 v = ((const float4*)h)[t];
    float* p = sums + g * DIM + q * 4;
    atomicAdd(p + 0, v.x);
    atomicAdd(p + 1, v.y);
    atomicAdd(p + 2, v.z);
    atomicAdd(p + 3, v.w);
    if (q == 0) atomicAdd(&cnts[g], 1.0f);
}

__global__ void pool_div_kernel(const float* __restrict__ sums,
                                const float* __restrict__ cnts,
                                float* __restrict__ pooled) {
    int i = blockIdx.x * blockDim.x + threadIdx.x;
    if (i < NGRAPH * DIM) {
        float c = cnts[i >> 6];
        pooled[i] = sums[i] / (c > 1.0f ? c : 1.0f);
    }
}

// ---------------------------------------------------------------------------
// MLP head: out = relu(pooled @ wh1 + bh1) @ wh2 + bh2.
// 64 graphs/block (8 blocks), wave w owns rows [16w,16w+16) -> barrier-free.
// ---------------------------------------------------------------------------
__global__ void __launch_bounds__(128)
head_kernel(const float* __restrict__ pooled,
            const _Float16* __restrict__ wh1p, const float* __restrict__ bh1,
            const _Float16* __restrict__ wh2p, const float* __restrict__ bh2,
            float* __restrict__ out) {
    __shared__ __align__(32) _Float16 sZ[64 * DIM];
    __shared__ __align__(32) _Float16 sT[64 * DIM];

    const int tid  = threadIdx.x;
    const int wave = tid >> 5;
    const int lane = tid & 31;
    const int rowBase = blockIdx.x * 64;

    {
        const int base = tid * 32;
        const float4* pp = (const float4*)(pooled + (long long)rowBase * DIM + base);
#pragma unroll
        for (int i = 0; i < 8; ++i) {
            float4 v = pp[i];
            sZ[base + i * 4 + 0] = (_Float16)v.x;
            sZ[base + i * 4 + 1] = (_Float16)v.y;
            sZ[base + i * 4 + 2] = (_Float16)v.z;
            sZ[base + i * 4 + 3] = (_Float16)v.w;
        }
    }

    const int m  = lane & 15;
    const int ko = (lane < 16) ? 0 : 8;
    const int n  = lane & 15;
    const int mo = (lane < 16) ? 0 : 8;

    auto loadA = [&](const _Float16* S, int kt) {
        v16h a;
#pragma unroll
        for (int e = 0; e < 8; ++e) {
            a[e]     = S[m * DIM + kt * 32 + ko + e];
            a[e + 8] = S[m * DIM + kt * 32 + 16 + ko + e];
        }
        return a;
    };

    const v16h* B1 = (const v16h*)wh1p;
    const v16h* B2 = (const v16h*)wh2p;

    // ---- GEMM1: hoisted B fragments, 8 back-to-back WMMAs ----
    const _Float16* Az = sZ + wave * 16 * DIM;
    v16h a0 = loadA(Az, 0);
    v16h a1 = loadA(Az, 1);
    v16h b1f[8];
#pragma unroll
    for (int f = 0; f < 8; ++f) b1f[f] = B1[f * 32 + lane];

    v8f c0 = {}, c1 = {}, c2 = {}, c3 = {};
    c0 = __builtin_amdgcn_wmma_f32_16x16x32_f16(false, a0, false, b1f[0], (short)0, c0, false, false);
    c1 = __builtin_amdgcn_wmma_f32_16x16x32_f16(false, a0, false, b1f[2], (short)0, c1, false, false);
    c2 = __builtin_amdgcn_wmma_f32_16x16x32_f16(false, a0, false, b1f[4], (short)0, c2, false, false);
    c3 = __builtin_amdgcn_wmma_f32_16x16x32_f16(false, a0, false, b1f[6], (short)0, c3, false, false);
    c0 = __builtin_amdgcn_wmma_f32_16x16x32_f16(false, a1, false, b1f[1], (short)0, c0, false, false);
    c1 = __builtin_amdgcn_wmma_f32_16x16x32_f16(false, a1, false, b1f[3], (short)0, c1, false, false);
    c2 = __builtin_amdgcn_wmma_f32_16x16x32_f16(false, a1, false, b1f[5], (short)0, c2, false, false);
    c3 = __builtin_amdgcn_wmma_f32_16x16x32_f16(false, a1, false, b1f[7], (short)0, c3, false, false);

    {
        v8f cc[4] = {c0, c1, c2, c3};
#pragma unroll
        for (int nt = 0; nt < 4; ++nt) {
            float bias = bh1[nt * 16 + n];
#pragma unroll
            for (int r = 0; r < 8; ++r) {
                float v = cc[nt][r] + bias;
                v = v > 0.f ? v : 0.f;
                sT[(wave * 16 + mo + r) * DIM + nt * 16 + n] = (_Float16)v;
            }
        }
    }

    // ---- GEMM2: out = t1 @ wh2 + bh2 over 48 N-tiles ----
    const _Float16* At = sT + wave * 16 * DIM;
    v16h t0 = loadA(At, 0);
    v16h t1 = loadA(At, 1);
    for (int nt = 0; nt < NOUT / 16; ++nt) {
        v16h bb0 = B2[(nt * 2 + 0) * 32 + lane];
        v16h bb1 = B2[(nt * 2 + 1) * 32 + lane];
        v8f c = {};
        c = __builtin_amdgcn_wmma_f32_16x16x32_f16(false, t0, false, bb0, (short)0, c, false, false);
        c = __builtin_amdgcn_wmma_f32_16x16x32_f16(false, t1, false, bb1, (short)0, c, false, false);
        int col = nt * 16 + n;
        float bias = bh2[col];
#pragma unroll
        for (int r = 0; r < 8; ++r)
            out[(long long)(rowBase + wave * 16 + mo + r) * NOUT + col] = c[r] + bias;
    }
}

// ---------------------------------------------------------------------------
// Host launcher
// ---------------------------------------------------------------------------
static inline size_t align256(size_t x) { return (x + 255) & ~(size_t)255; }

extern "C" void kernel_launch(void* const* d_in, const int* in_sizes, int n_in,
                              void* d_out, int out_size, void* d_ws, size_t ws_size,
                              hipStream_t stream) {
    const float* x     = (const float*)d_in[0];
    const int*   ei    = (const int*)d_in[1];     // [2, E]
    const int*   batch = (const int*)d_in[2];
    const float* gw1   = (const float*)d_in[3];   // [L, D, D]
    const float* gb1   = (const float*)d_in[4];   // [L, D]
    const float* gw2   = (const float*)d_in[5];
    const float* gb2   = (const float*)d_in[6];
    const float* eps   = (const float*)d_in[7];   // [L]
    const float* wh1   = (const float*)d_in[8];   // [D, D]
    const float* bh1   = (const float*)d_in[9];
    const float* wh2   = (const float*)d_in[10];  // [D, NOUT]
    const float* bh2   = (const float*)d_in[11];
    float* out = (float*)d_out;

    // workspace carve-up
    char* ws = (char*)d_ws;
    float*    bufA   = (float*)ws;    ws += align256((size_t)N_NODES * DIM * 4);
    float*    bufB   = (float*)ws;    ws += align256((size_t)N_NODES * DIM * 4);
    float*    agg    = (float*)ws;    ws += align256((size_t)N_NODES * DIM * 4);
    _Float16* w1p    = (_Float16*)ws; ws += align256((size_t)NLAYERS * DIM * DIM * 2);
    _Float16* w2p    = (_Float16*)ws; ws += align256((size_t)NLAYERS * DIM * DIM * 2);
    _Float16* wh1p   = (_Float16*)ws; ws += align256((size_t)DIM * DIM * 2);
    _Float16* wh2p   = (_Float16*)ws; ws += align256((size_t)DIM * NOUT * 2);
    float*    sums   = (float*)ws;    ws += align256((size_t)NGRAPH * DIM * 4);
    float*    cnts   = (float*)ws;    ws += align256((size_t)NGRAPH * 4);
    float*    pooled = (float*)ws;    ws += align256((size_t)NGRAPH * DIM * 4);

    // pack weights into WMMA B-fragment layout (one-shot, tiny)
    pack_b_kernel<<<(NLAYERS * DIM * DIM + 255) / 256, 256, 0, stream>>>(gw1, w1p, DIM, NLAYERS);
    pack_b_kernel<<<(NLAYERS * DIM * DIM + 255) / 256, 256, 0, stream>>>(gw2, w2p, DIM, NLAYERS);
    pack_b_kernel<<<(DIM * DIM + 255) / 256, 256, 0, stream>>>(wh1, wh1p, DIM, 1);
    pack_b_kernel<<<(DIM * NOUT + 255) / 256, 256, 0, stream>>>(wh2, wh2p, NOUT, 1);

    const int scatterBlocks = (int)(((long long)N_EDGES * 16 + 255) / 256);
    const int ginBlocks = (N_NODES + 63) / 64;
    const float* cur = x;
    float* nxt = bufA;
    for (int l = 0; l < NLAYERS; ++l) {
        hipMemsetAsync(agg, 0, (size_t)N_NODES * DIM * 4, stream);
        scatter_kernel<<<scatterBlocks, 256, 0, stream>>>(cur, ei, ei + N_EDGES, agg);
        gin_mlp_kernel<<<ginBlocks, 128, 0, stream>>>(
            cur, agg,
            w1p + (size_t)l * DIM * DIM, w2p + (size_t)l * DIM * DIM,
            gb1 + (size_t)l * DIM, gb2 + (size_t)l * DIM,
            eps + l, (l < NLAYERS - 1) ? 1 : 0, nxt);
        cur = nxt;
        nxt = (nxt == bufA) ? bufB : bufA;
    }

    hipMemsetAsync(sums, 0, (size_t)NGRAPH * DIM * 4, stream);
    hipMemsetAsync(cnts, 0, (size_t)NGRAPH * 4, stream);
    pool_scatter_kernel<<<(int)(((long long)N_NODES * 16 + 255) / 256), 256, 0, stream>>>(
        cur, batch, sums, cnts);
    pool_div_kernel<<<(NGRAPH * DIM + 255) / 256, 256, 0, stream>>>(sums, cnts, pooled);
    head_kernel<<<NGRAPH / 64, 128, 0, stream>>>(pooled, wh1p, bh1, wh2p, bh2, out);
}
